// WindowAttention_54743653155451
// MI455X (gfx1250) — matrix operand order repaired and checked by reading, not compiled, not run on hardware
//
#include <hip/hip_runtime.h>
#include <hip/hip_bf16.h>

typedef _Float16 h16;
typedef __attribute__((ext_vector_type(16))) _Float16 v16h;
typedef __attribute__((ext_vector_type(8)))  float    v8f;
typedef __attribute__((ext_vector_type(4)))  int      v4i;

#define NTOK   64
#define CDIM   128
#define NHEAD  4
#define HDIM   32
#define HID    32
#define NWIN   16384
#define GRID   4096   // persistent blocks; each handles NWIN/GRID windows
#define XP     130    // padded pitch (halves) for 128-wide LDS tiles
#define SPP    66     // padded pitch for 64-wide score tiles
#define HP     34     // padded pitch for 32-wide hidden tiles

// gfx1250 async global->LDS path (probe via __has_builtin; falls back to
// synchronous loads + global_prefetch_b8 when absent / on the host pass)
#if defined(__HIP_DEVICE_COMPILE__) && __has_builtin(__builtin_amdgcn_global_load_async_to_lds_b128)
#define HAVE_ASYNC 1
#else
#define HAVE_ASYNC 0
#endif

union FragU { v16h v; h16 h[16]; unsigned int u[8]; };

// ---- fragment gathers (this code's k order: k = 16*(lane/16) + s in both A and B) ----
__device__ __forceinline__ v16h frag_A(const h16* base, int row0, int pitch, int k0, int lane) {
  FragU f;
  const int m  = lane & 15;
  const int kh = (lane >> 4) << 4;
  const h16* rp = base + (row0 + m) * pitch + k0;
#pragma unroll
  for (int v = 0; v < 8; ++v) {
    int k = kh + (v << 1);
    f.u[v] = *(const unsigned int*)(rp + k);   // two consecutive halves
  }
  return f.v;
}
__device__ __forceinline__ v16h frag_B_contig(const h16* base, int n0, int ldk, int k0, int lane) {
  FragU f;
  const int n    = lane & 15;
  const int koff = (lane >> 4) << 4;
  const h16* p = base + (n0 + n) * ldk + k0 + koff;
#pragma unroll
  for (int v = 0; v < 8; ++v) f.u[v] = *(const unsigned int*)(p + (v << 1));
  return f.v;
}
__device__ __forceinline__ v16h frag_B_strided(const h16* base, int n0, int nstride,
                                               int k0, int kstride, int lane) {
  FragU f;
  const int n    = lane & 15;
  const int koff = (lane >> 4) << 4;
  const h16* p = base + (n0 + n) * nstride + (k0 + koff) * kstride;
#pragma unroll
  for (int v = 0; v < 8; ++v) {
    f.h[2 * v]     = p[(2 * v) * kstride];
    f.h[2 * v + 1] = p[(2 * v + 1) * kstride];
  }
  return f.v;
}

__device__ __forceinline__ v8f wmma_f16(v16h a, v16h b, v8f c) {
  return __builtin_amdgcn_wmma_f32_16x16x32_f16(false, a, false, b, (short)0, c, false, false);
}

#if HAVE_ASYNC
// Exact pointer types per the clang diagnostic: v4i in global AS (printed as
// "__device__", AS1) and LDS AS (AS3). Cast via integers: AS1 == generic
// global address; AS3 == low 32 bits of the generic LDS address (flat
// aperture: LDS offset is the address's low 32 bits).
typedef __attribute__((address_space(1))) v4i g_v4i;
typedef __attribute__((address_space(3))) v4i l_v4i;
__device__ __forceinline__ g_v4i* as_glb(const void* p) {
  return (g_v4i*)(unsigned long long)p;
}
__device__ __forceinline__ l_v4i* as_lds(void* p) {
  return (l_v4i*)(unsigned int)(unsigned long long)p;
}
__device__ __forceinline__ void issue_x_async(float* xraw_s, const float* xg, int tid) {
#pragma unroll
  for (int i = 0; i < 8; ++i) {
    int off = i * 1024 + tid * 4;   // floats; 256 lanes x 16B per op
    __builtin_amdgcn_global_load_async_to_lds_b128(as_glb(xg + off), as_lds(xraw_s + off), 0, 0);
  }
}
__device__ __forceinline__ void wait_async0() {
#if __has_builtin(__builtin_amdgcn_s_wait_asynccnt)
  __builtin_amdgcn_s_wait_asynccnt(0);
#else
  asm volatile("s_wait_asynccnt 0" ::: "memory");
#endif
}
#endif

// ---- prep: fp32 weights -> f16 transposed (K contiguous), bias table [h][q][k] ----
__global__ void prep_weights(const float* __restrict__ wq,  const float* __restrict__ wkv,
                             const float* __restrict__ fc1w, const float* __restrict__ fc2w,
                             const float* __restrict__ rpb,  const int* __restrict__ rel_idx,
                             h16* __restrict__ wqT, h16* __restrict__ wkvT,
                             h16* __restrict__ fc1T, h16* __restrict__ fc2T,
                             float* __restrict__ biasT) {
  int t = blockIdx.x * blockDim.x + threadIdx.x;
  if (t < 16384) {                                   // wqT[n*128+k] = wq[k*128+n]
    int n = t >> 7, k = t & 127;
    wqT[t] = (h16)wq[k * 128 + n];
  } else if (t < 16384 + 32768) {                    // wkvT[n*128+k] = wkv[k*256+n]
    int u = t - 16384; int n = u >> 7, k = u & 127;
    wkvT[u] = (h16)wkv[k * 256 + n];
  } else if (t < 16384 + 32768 + 4096) {             // fc1T[n*128+k] = fc1[k*32+n]
    int u = t - 49152; int n = u >> 7, k = u & 127;
    fc1T[u] = (h16)fc1w[k * 32 + n];
  } else if (t < 16384 + 32768 + 4096 + 4096) {      // fc2T[n*32+k] = fc2[k*128+n]
    int u = t - 53248; int n = u >> 5, k = u & 31;
    fc2T[u] = (h16)fc2w[k * 128 + n];
  } else if (t < 16384 + 32768 + 4096 + 4096 + 16384) {
    int u = t - 57344; int h = u >> 12; int qk = u & 4095;
    biasT[u] = rpb[rel_idx[qk] * NHEAD + h];         // [h][q][k]
  }
}

// ---- fused window attention + MLP: persistent blocks, 8 wave32s each ----
__global__ void __launch_bounds__(256)
winattn_kernel(const float* __restrict__ x,
               const float* __restrict__ bq,  const float* __restrict__ bkv,
               const float* __restrict__ fc1b, const float* __restrict__ fc2b,
               const h16* __restrict__ wqT,  const h16* __restrict__ wkvT,
               const h16* __restrict__ fc1T, const h16* __restrict__ fc2T,
               const float* __restrict__ biasT,
               float* __restrict__ out) {
  __shared__ h16   xs[NTOK * XP];           // x (f16); reused as attention output
  __shared__ h16   qs[NTOK * XP];           // Q*scale; reused as MLP hidden
  __shared__ h16   ks[NTOK * XP];
  __shared__ h16   vs[NTOK * XP];
  __shared__ h16   sP[NHEAD * NTOK * SPP];  // scores, then probs (in place)
  __shared__ float xraw[NTOK * CDIM];       // raw f32 stage (async target)
  h16* os = xs;                             // alias: x dead after projections
  h16* hs = qs;                             // alias: Q dead after scores

  const int tid  = threadIdx.x;
  const int wave = tid >> 5;
  const int lane = tid & 31;
  const int nloc = lane & 15;
  const int rofs = (lane >> 4) << 3;
  const float qscale = 0.17677669529663687f;   // 1/sqrt(32)

  long long win = blockIdx.x;
#if HAVE_ASYNC
  if (win < NWIN) issue_x_async(xraw, x + win * (NTOK * CDIM), tid);
#endif

  for (; win < NWIN; win += GRID) {
    const long long nxt = win + GRID;

#if HAVE_ASYNC
    // Phase A: wait for this window's async copy, convert f32 -> f16 into xs,
    // then immediately kick off the next window's copy (overlaps all compute).
    wait_async0();
    __syncthreads();                     // also orders vs. prev iter phases E/F
#pragma unroll
    for (int i = 0; i < 8; ++i) {
      int idx = i * 1024 + tid * 4;
      float4 v = *(const float4*)(xraw + idx);
      int row = idx >> 7, col = idx & 127;
      h16* d = xs + row * XP + col;
      d[0] = (h16)v.x; d[1] = (h16)v.y; d[2] = (h16)v.z; d[3] = (h16)v.w;
    }
    __syncthreads();                     // convert done; xraw free for reuse
    if (nxt < NWIN) issue_x_async(xraw, x + nxt * (NTOK * CDIM), tid);
#else
    // Fallback: prefetch next window, synchronous vec4 load of this one.
    if (nxt < NWIN) __builtin_prefetch(x + nxt * (NTOK * CDIM) + tid * 32, 0, 3);
    const float* xg = x + win * (NTOK * CDIM);
    __syncthreads();                     // order vs. prev iter phases E/F
#pragma unroll
    for (int i = 0; i < 8; ++i) {
      int idx = i * 1024 + tid * 4;
      float4 v = *(const float4*)(xg + idx);
      int row = idx >> 7, col = idx & 127;
      h16* d = xs + row * XP + col;
      d[0] = (h16)v.x; d[1] = (h16)v.y; d[2] = (h16)v.z; d[3] = (h16)v.w;
    }
    __syncthreads();
#endif

    // Phase B: [Q|K|V] = x @ [wq|wkv] + bias   (96 tiles 16x16, K=128)
    for (int t = wave; t < 96; t += 8) {
      int mt = t & 3, nt = t >> 2;
      v8f acc = {};
#pragma unroll
      for (int kt = 0; kt < 4; ++kt) {
        v16h a = frag_A(xs, mt * 16, XP, kt * 32, lane);
        v16h b = (nt < 8) ? frag_B_contig(wqT, nt * 16, 128, kt * 32, lane)
                          : frag_B_contig(wkvT, (nt - 8) * 16, 128, kt * 32, lane);
        acc = wmma_f16(a, b, acc);
      }
      int nglob = nt * 16 + nloc;
      h16* dst; int dcol; float bias; float sc = 1.0f;
      if (nglob < 128)      { bias = bq[nglob];  dst = qs; dcol = nglob; sc = qscale; }
      else { int j = nglob - 128; bias = bkv[j];
             if (j < 128)   { dst = ks; dcol = j; }
             else           { dst = vs; dcol = j - 128; } }
#pragma unroll
      for (int r = 0; r < 8; ++r)
        dst[(mt * 16 + rofs + r) * XP + dcol] = (h16)((acc[r] + bias) * sc);
    }
    __syncthreads();

    // Phase C: S = (Q*scale) @ K^T + relpos bias   (per head 64x64, K=32)
    for (int t = wave; t < 64; t += 8) {
      int h = t >> 4, rem = t & 15, mt = rem >> 2, nt = rem & 3;
      v16h a = frag_A(qs + h * HDIM, mt * 16, XP, 0, lane);
      v16h b = frag_B_contig(ks + h * HDIM, nt * 16, XP, 0, lane); // B[d][tok]
      v8f acc = {};
      acc = wmma_f16(a, b, acc);
      int kcol = nt * 16 + nloc;
      const float* bb = biasT + h * 4096 + kcol;
#pragma unroll
      for (int r = 0; r < 8; ++r) {
        int qrow = mt * 16 + rofs + r;
        sP[(h * 64 + qrow) * SPP + kcol] = (h16)(acc[r] + bb[qrow * 64]);
      }
    }
    __syncthreads();

    // Softmax: 4 heads x 64 rows == 256 threads, one row each, in place
    {
      int h = tid >> 6, row = tid & 63;
      h16* p = sP + (h * 64 + row) * SPP;
      float m = -1e30f;
      for (int j = 0; j < 64; ++j) m = fmaxf(m, (float)p[j]);
      float s = 0.f;
      for (int j = 0; j < 64; ++j) s += __expf((float)p[j] - m);
      float inv = 1.0f / s;
      for (int j = 0; j < 64; ++j) p[j] = (h16)(__expf((float)p[j] - m) * inv);
    }
    __syncthreads();

    // Phase D: O = P @ V   (per head 64x32, K=64)
    for (int t = wave; t < 32; t += 8) {
      int h = t >> 3, rem = t & 7, mt = rem >> 1, nt = rem & 1;
      v8f acc = {};
#pragma unroll
      for (int kt = 0; kt < 2; ++kt) {
        v16h a = frag_A(sP + h * 64 * SPP, mt * 16, SPP, kt * 32, lane);
        v16h b = frag_B_strided(vs + h * HDIM + nt * 16, 0, 1, kt * 32, XP, lane);
        acc = wmma_f16(a, b, acc);
      }
      int col = h * HDIM + nt * 16 + nloc;
#pragma unroll
      for (int r = 0; r < 8; ++r)
        os[(mt * 16 + rofs + r) * XP + col] = (h16)acc[r];
    }
    __syncthreads();

    // Phase E: hidden = GELU(O @ fc1 + b1)   (64x32, K=128) — one tile per wave
    {
      int mt = wave & 3, nt = wave >> 2;
      v8f acc = {};
#pragma unroll
      for (int kt = 0; kt < 4; ++kt) {
        v16h a = frag_A(os, mt * 16, XP, kt * 32, lane);
        v16h b = frag_B_contig(fc1T, nt * 16, 128, kt * 32, lane);
        acc = wmma_f16(a, b, acc);
      }
      int col = nt * 16 + nloc;
      float bias = fc1b[col];
#pragma unroll
      for (int r = 0; r < 8; ++r) {
        float u = acc[r] + bias;
        float g = 0.5f * u * (1.0f + erff(u * 0.70710678118654752f));  // exact GELU
        hs[(mt * 16 + rofs + r) * HP + col] = (h16)g;
      }
    }
    __syncthreads();

    // Phase F: y = hidden @ fc2 + b2   (64x128, K=32) — stream f32 to global
    float* og = out + win * (NTOK * CDIM);
    for (int t = wave; t < 32; t += 8) {
      int mt = t & 3, nt = t >> 2;
      v16h a = frag_A(hs, mt * 16, HP, 0, lane);
      v16h b = frag_B_contig(fc2T, nt * 16, 32, 0, lane);
      v8f acc = {};
      acc = wmma_f16(a, b, acc);
      int col = nt * 16 + nloc;
      float bias = fc2b[col];
#pragma unroll
      for (int r = 0; r < 8; ++r)
        og[(mt * 16 + rofs + r) * CDIM + col] = acc[r] + bias;
    }
    // next iteration's top-of-loop barrier orders xs/qs reuse vs. phases E/F
  }
}

extern "C" void kernel_launch(void* const* d_in, const int* in_sizes, int n_in,
                              void* d_out, int out_size, void* d_ws, size_t ws_size,
                              hipStream_t stream) {
  const float* x    = (const float*)d_in[0];
  const float* rpb  = (const float*)d_in[1];
  const float* wq   = (const float*)d_in[2];
  const float* bq   = (const float*)d_in[3];
  const float* wkv  = (const float*)d_in[4];
  const float* bkv  = (const float*)d_in[5];
  const float* fc1w = (const float*)d_in[6];
  const float* fc1b = (const float*)d_in[7];
  const float* fc2w = (const float*)d_in[8];
  const float* fc2b = (const float*)d_in[9];
  const int*   ridx = (const int*)d_in[10];

  char* ws = (char*)d_ws;
  h16*   wqT   = (h16*)(ws);                                 // 16384 halves
  h16*   wkvT  = (h16*)(ws + 32768);                         // 32768 halves
  h16*   fc1T  = (h16*)(ws + 32768 + 65536);                 //  4096 halves
  h16*   fc2T  = (h16*)(ws + 32768 + 65536 + 8192);          //  4096 halves
  float* biasT = (float*)(ws + 32768 + 65536 + 8192 + 8192); // 16384 floats
  (void)in_sizes; (void)n_in; (void)out_size; (void)ws_size;

  prep_weights<<<288, 256, 0, stream>>>(wq, wkv, fc1w, fc2w, rpb, ridx,
                                        wqT, wkvT, fc1T, fc2T, biasT);
  winattn_kernel<<<GRID, 256, 0, stream>>>(x, bq, bkv, fc1b, fc2b,
                                           wqT, wkvT, fc1T, fc2T, biasT,
                                           (float*)d_out);
}
